// NeuroGenderTokenizer_47596827574733
// MI455X (gfx1250) — matrix operand order repaired and checked by reading, not compiled, not run on hardware
//
#include <hip/hip_runtime.h>
#include <hip/hip_bf16.h>

// ---------------------------------------------------------------------------
// GCN encoder (VGAE) for MI455X / gfx1250.
// Dense projections run on v_wmma_f32_16x16x32_bf16 (f32 accumulate),
// register-blocked 16x64 per wave (4 WMMA tiles sharing one A fragment).
// Sparse aggregation via global_atomic_add_f32.
// ---------------------------------------------------------------------------

typedef __attribute__((ext_vector_type(16))) __bf16 v16bf;
typedef __attribute__((ext_vector_type(8)))  float  v8f;

#define N_NODES 10000
#define N_EDGES 160000
#define F0 128
#define F1 1024
#define F2 2000
#define F2P 2016   // F2 padded to a multiple of 32 for WMMA K loop
#define F3 1024

static __device__ __forceinline__ unsigned short f2bf_rne(float x) {
  unsigned int u = __float_as_uint(x);
  unsigned int r = (u + 0x7FFFu + ((u >> 16) & 1u)) >> 16;  // round-nearest-even
  return (unsigned short)r;
}

// ---------------------------- degree / norm --------------------------------

__global__ void k_deg_init(float* deg, int n) {
  int i = blockIdx.x * blockDim.x + threadIdx.x;
  if (i < n) deg[i] = 1.0f;  // self loop
}

__global__ void k_deg_count(const long long* __restrict__ dst, float* deg, int e) {
  int i = blockIdx.x * blockDim.x + threadIdx.x;
  if (i < e) atomicAdd(&deg[(int)dst[i]], 1.0f);
}

__global__ void k_dinv(const float* __restrict__ deg, float* dinv, int n) {
  int i = blockIdx.x * blockDim.x + threadIdx.x;
  if (i < n) dinv[i] = rsqrtf(deg[i]);  // deg >= 1 always
}

// ------------------------ f32 -> bf16 (opt. ReLU, K pad) -------------------

__global__ void k_f32_to_bf16(const float* __restrict__ in, unsigned short* __restrict__ out,
                              int rows, int f, int fpad, int do_relu) {
  int idx = blockIdx.x * blockDim.x + threadIdx.x;
  int total = rows * fpad;
  if (idx >= total) return;
  int r = idx / fpad, c = idx - r * fpad;
  float v = 0.0f;
  if (c < f) {
    v = in[(size_t)r * f + c];
    if (do_relu) v = fmaxf(v, 0.0f);
  }
  out[idx] = f2bf_rne(v);
}

// ----------------- pack W[K,N] f32 -> WMMA B-fragment bf16 -----------------
// Layout per CDNA5 ISA 7.12.2 (16-bit B, 32x16): lane%16 = N column within the
// 16-wide tile; element e of lane half h holds K=8h+e for e<8 and
// K=16+8h+(e-8) for e>=8 (matches the A fragment order).
// Packed index: ((ntile*Ktiles + ktile)*32 + lane)*16 + e.

__global__ void k_pack_w(const float* __restrict__ w, unsigned short* __restrict__ bp,
                         int k, int n, int kpad) {
  int ktiles = kpad >> 5;
  int idx = blockIdx.x * blockDim.x + threadIdx.x;
  int total = (n >> 4) * ktiles * 512;
  if (idx >= total) return;
  int e    = idx & 15;
  int lane = (idx >> 4) & 31;
  int kt   = (idx >> 9) % ktiles;
  int nt   = idx / (ktiles << 9);
  int kk = (kt << 5) + ((lane & 16) ? 8 : 0) + (e & 7) + ((e & 8) ? 16 : 0);
  int nn = (nt << 4) + (lane & 15);
  float v = (kk < k) ? w[(size_t)kk * n + nn] : 0.0f;  // zero-pad K remainder
  bp[idx] = f2bf_rne(v);
}

// --------------------------- WMMA bf16 GEMM --------------------------------
// C[m,n] = A[m,kpad](bf16) * Bpacked (bf16), f32 accumulate.
// Block = 8 waves = 128x64 output tile; each wave computes 16x64 (4 WMMA
// tiles) reusing a single A fragment per K-step -> 4 WMMAs per A load.

__global__ __launch_bounds__(256) void k_gemm_bf16_wmma(
    const unsigned short* __restrict__ a, const unsigned short* __restrict__ bp,
    float* __restrict__ c, int m, int n, int kpad) {
  const int lane = threadIdx.x & 31;
  const int wave = threadIdx.x >> 5;
  const int m0 = (blockIdx.y * 8 + wave) * 16;
  if (m0 >= m) return;  // wave-uniform: EXEC stays all-ones for WMMA
  const int n0 = blockIdx.x * 64;

  const int ktiles = kpad >> 5;
  const int ntiles = n >> 4;  // n is a multiple of 16
  // A fragment: lane%16 = row, lanes>=16 take K offsets +8 in each half.
  const int kbase = (lane & 16) ? 8 : 0;
  const unsigned short* ap = a + (size_t)(m0 + (lane & 15)) * kpad + kbase;
  // B fragments: fully packed, contiguous 32 B per lane per K tile.
  // Clamp tile indices for the partial 64-wide tile (N=2000): redundant loads,
  // results discarded at store; avoids OOB, keeps EXEC uniform.
  const int ntb = n0 >> 4;
  const int nt0 = (ntb + 0 < ntiles) ? ntb + 0 : ntiles - 1;
  const int nt1 = (ntb + 1 < ntiles) ? ntb + 1 : ntiles - 1;
  const int nt2 = (ntb + 2 < ntiles) ? ntb + 2 : ntiles - 1;
  const int nt3 = (ntb + 3 < ntiles) ? ntb + 3 : ntiles - 1;
  const unsigned short* bpp0 = bp + ((size_t)nt0 * ktiles * 32 + lane) * 16;
  const unsigned short* bpp1 = bp + ((size_t)nt1 * ktiles * 32 + lane) * 16;
  const unsigned short* bpp2 = bp + ((size_t)nt2 * ktiles * 32 + lane) * 16;
  const unsigned short* bpp3 = bp + ((size_t)nt3 * ktiles * 32 + lane) * 16;

  v8f acc0 = {}, acc1 = {}, acc2 = {}, acc3 = {};
  for (int kt = 0; kt < ktiles; ++kt) {
    const unsigned int* ap32 = (const unsigned int*)(ap + kt * 32);
    __builtin_prefetch(ap + (kt + 1) * 32, 0, 0);  // global_prefetch_b8
    union { unsigned int u[8]; v16bf v; } af;
    af.u[0] = ap32[0];   // K+{0,1}
    af.u[1] = ap32[1];   // K+{2,3}
    af.u[2] = ap32[2];   // K+{4,5}
    af.u[3] = ap32[3];   // K+{6,7}
    af.u[4] = ap32[8];   // K+{16,17}
    af.u[5] = ap32[9];   // K+{18,19}
    af.u[6] = ap32[10];  // K+{20,21}
    af.u[7] = ap32[11];  // K+{22,23}
    union { uint4 q[2]; v16bf v; } b0, b1, b2, b3;
    const uint4* q0 = (const uint4*)(bpp0 + (size_t)kt * 512);
    const uint4* q1 = (const uint4*)(bpp1 + (size_t)kt * 512);
    const uint4* q2 = (const uint4*)(bpp2 + (size_t)kt * 512);
    const uint4* q3 = (const uint4*)(bpp3 + (size_t)kt * 512);
    b0.q[0] = q0[0]; b0.q[1] = q0[1];
    b1.q[0] = q1[0]; b1.q[1] = q1[1];
    b2.q[0] = q2[0]; b2.q[1] = q2[1];
    b3.q[0] = q3[0]; b3.q[1] = q3[1];
    acc0 = __builtin_amdgcn_wmma_f32_16x16x32_bf16(false, af.v, false, b0.v, (short)0, acc0, false, false);
    acc1 = __builtin_amdgcn_wmma_f32_16x16x32_bf16(false, af.v, false, b1.v, (short)0, acc1, false, false);
    acc2 = __builtin_amdgcn_wmma_f32_16x16x32_bf16(false, af.v, false, b2.v, (short)0, acc2, false, false);
    acc3 = __builtin_amdgcn_wmma_f32_16x16x32_bf16(false, af.v, false, b3.v, (short)0, acc3, false, false);
  }

  // C/D layout: lane%16 = N; VGPR r -> M = r (+8 for lanes 16-31).
  const int cn = lane & 15;
  const int cm = m0 + ((lane >> 4) << 3);
  float* cp = c + (size_t)cm * n + n0 + cn;
  if (n0 + 0 * 16 < n) {
#pragma unroll
    for (int r = 0; r < 8; ++r) cp[(size_t)r * n + 0] = acc0[r];
  }
  if (n0 + 1 * 16 < n) {
#pragma unroll
    for (int r = 0; r < 8; ++r) cp[(size_t)r * n + 16] = acc1[r];
  }
  if (n0 + 2 * 16 < n) {
#pragma unroll
    for (int r = 0; r < 8; ++r) cp[(size_t)r * n + 32] = acc2[r];
  }
  if (n0 + 3 * 16 < n) {
#pragma unroll
    for (int r = 0; r < 8; ++r) cp[(size_t)r * n + 48] = acc3[r];
  }
}

// -------------------------- sparse aggregation -----------------------------

// out[i,f] = h[i,f]*dinv[i]^2 + bias[f]   (self-loop term + bias; full overwrite)
__global__ void k_agg_init(const float* __restrict__ h, const float* __restrict__ dinv,
                           const float* __restrict__ bias, float* __restrict__ out,
                           int nnodes, int f) {
  int idx = blockIdx.x * blockDim.x + threadIdx.x;
  int total = nnodes * f;
  if (idx >= total) return;
  int i = idx / f, c = idx - i * f;
  float di = dinv[i];
  out[idx] = h[idx] * di * di + bias[c];
}

// out[dst] += dinv[src]*dinv[dst] * h[src]  (4 features per thread)
__global__ void k_agg_edges(const float* __restrict__ h, const long long* __restrict__ src,
                            const long long* __restrict__ dst, const float* __restrict__ dinv,
                            float* __restrict__ out, int e, int f) {
  int idx = blockIdx.x * blockDim.x + threadIdx.x;
  int fq = f >> 2;
  int total = e * fq;
  if (idx >= total) return;
  int ei = idx / fq;
  int c = (idx - ei * fq) << 2;
  int s = (int)src[ei], d = (int)dst[ei];
  float w = dinv[s] * dinv[d];
  const float* hp = h + (size_t)s * f + c;
  float* op = out + (size_t)d * f + c;
  atomicAdd(&op[0], w * hp[0]);
  atomicAdd(&op[1], w * hp[1]);
  atomicAdd(&op[2], w * hp[2]);
  atomicAdd(&op[3], w * hp[3]);
}

// ---------------------------------------------------------------------------

static inline size_t align_up(size_t x) { return (x + 255) & ~(size_t)255; }
static inline int cdiv(int a, int b) { return (a + b - 1) / b; }

extern "C" void kernel_launch(void* const* d_in, const int* in_sizes, int n_in,
                              void* d_out, int out_size, void* d_ws, size_t ws_size,
                              hipStream_t stream) {
  (void)in_sizes; (void)n_in; (void)out_size; (void)ws_size;
  const float*     x0    = (const float*)d_in[0];
  const long long* edges = (const long long*)d_in[1];
  const long long* src   = edges;
  const long long* dst   = edges + N_EDGES;
  const float* W1   = (const float*)d_in[2]; const float* b1   = (const float*)d_in[3];
  const float* W2   = (const float*)d_in[4]; const float* b2   = (const float*)d_in[5];
  const float* Wmu  = (const float*)d_in[6]; const float* bmu  = (const float*)d_in[7];
  const float* Wstd = (const float*)d_in[8]; const float* bstd = (const float*)d_in[9];
  float* out_mu  = (float*)d_out;
  float* out_std = out_mu + (size_t)N_NODES * F3;

  // ---- workspace arena ----
  char* ws = (char*)d_ws; size_t off = 0;
  auto carve = [&](size_t bytes) { char* p = ws + off; off += align_up(bytes); return p; };
  float*          deg   = (float*)carve((size_t)N_NODES * 4);
  float*          dinv  = (float*)carve((size_t)N_NODES * 4);
  unsigned short* x0b   = (unsigned short*)carve((size_t)N_NODES * F0 * 2);
  unsigned short* w1p   = (unsigned short*)carve((size_t)F0 * F1 * 2);
  unsigned short* w2p   = (unsigned short*)carve((size_t)F1 * F2 * 2);
  unsigned short* wmup  = (unsigned short*)carve((size_t)F2P * F3 * 2);
  unsigned short* wstdp = (unsigned short*)carve((size_t)F2P * F3 * 2);
  // R1: GEMM outputs (h1 -> h2 -> hmu+hstd, sequentially reused)
  float* R1 = (float*)carve((size_t)2 * N_NODES * F3 * 4);  // 81.92 MB
  // R2: aggregation accumulator (x1acc -> x2acc)
  float* R2 = (float*)carve((size_t)N_NODES * F2 * 4);      // 80 MB
  // R3: bf16 activations (x1b -> x2b)
  unsigned short* R3 = (unsigned short*)carve((size_t)N_NODES * F2P * 2);  // 40.3 MB

  float* h1   = R1;
  float* h2   = R1;
  float* hmu  = R1;
  float* hstd = R1 + (size_t)N_NODES * F3;

  const int TB = 256;

  // ---- precompute: bf16 input, packed weights, degrees ----
  k_f32_to_bf16<<<cdiv(N_NODES * F0, TB), TB, 0, stream>>>(x0, x0b, N_NODES, F0, F0, 0);
  k_pack_w<<<cdiv((F1 / 16) * (F0 / 32) * 512, TB), TB, 0, stream>>>(W1, w1p, F0, F1, F0);
  k_pack_w<<<cdiv((F2 / 16) * (F1 / 32) * 512, TB), TB, 0, stream>>>(W2, w2p, F1, F2, F1);
  k_pack_w<<<cdiv((F3 / 16) * (F2P / 32) * 512, TB), TB, 0, stream>>>(Wmu, wmup, F2, F3, F2P);
  k_pack_w<<<cdiv((F3 / 16) * (F2P / 32) * 512, TB), TB, 0, stream>>>(Wstd, wstdp, F2, F3, F2P);
  k_deg_init<<<cdiv(N_NODES, TB), TB, 0, stream>>>(deg, N_NODES);
  k_deg_count<<<cdiv(N_EDGES, TB), TB, 0, stream>>>(dst, deg, N_EDGES);
  k_dinv<<<cdiv(N_NODES, TB), TB, 0, stream>>>(deg, dinv, N_NODES);

  dim3 blk(TB);

  // ---- layer 1: h1 = x0 @ W1 ; x1 = relu(agg(h1) + b1) ----
  { dim3 g(cdiv(F1, 64), cdiv(N_NODES, 128));
    k_gemm_bf16_wmma<<<g, blk, 0, stream>>>(x0b, w1p, h1, N_NODES, F1, F0); }
  k_agg_init<<<cdiv(N_NODES * F1, TB), TB, 0, stream>>>(h1, dinv, b1, R2, N_NODES, F1);
  k_agg_edges<<<cdiv(N_EDGES * (F1 / 4), TB), TB, 0, stream>>>(h1, src, dst, dinv, R2, N_EDGES, F1);
  k_f32_to_bf16<<<cdiv(N_NODES * F1, TB), TB, 0, stream>>>(R2, R3, N_NODES, F1, F1, 1);

  // ---- layer 2: h2 = x1 @ W2 ; x2 = relu(agg(h2) + b2) ----
  { dim3 g(cdiv(F2, 64), cdiv(N_NODES, 128));
    k_gemm_bf16_wmma<<<g, blk, 0, stream>>>(R3, w2p, h2, N_NODES, F2, F1); }
  k_agg_init<<<cdiv(N_NODES * F2, TB), TB, 0, stream>>>(h2, dinv, b2, R2, N_NODES, F2);
  k_agg_edges<<<cdiv(N_EDGES * (F2 / 4), TB), TB, 0, stream>>>(h2, src, dst, dinv, R2, N_EDGES, F2);
  k_f32_to_bf16<<<cdiv(N_NODES * F2P, TB), TB, 0, stream>>>(R2, R3, N_NODES, F2, F2P, 1);

  // ---- mu / std heads: project then aggregate into d_out ----
  { dim3 g(cdiv(F3, 64), cdiv(N_NODES, 128));
    k_gemm_bf16_wmma<<<g, blk, 0, stream>>>(R3, wmup, hmu, N_NODES, F3, F2P);
    k_gemm_bf16_wmma<<<g, blk, 0, stream>>>(R3, wstdp, hstd, N_NODES, F3, F2P); }
  k_agg_init<<<cdiv(N_NODES * F3, TB), TB, 0, stream>>>(hmu, dinv, bmu, out_mu, N_NODES, F3);
  k_agg_edges<<<cdiv(N_EDGES * (F3 / 4), TB), TB, 0, stream>>>(hmu, src, dst, dinv, out_mu, N_EDGES, F3);
  k_agg_init<<<cdiv(N_NODES * F3, TB), TB, 0, stream>>>(hstd, dinv, bstd, out_std, N_NODES, F3);
  k_agg_edges<<<cdiv(N_EDGES * (F3 / 4), TB), TB, 0, stream>>>(hstd, src, dst, dinv, out_std, N_EDGES, F3);
}
